// CrossAttentionGate_41815801593918
// MI455X (gfx1250) — matrix operand order repaired and checked by reading, not compiled, not run on hardware
//
#include <hip/hip_runtime.h>

// ---------------------------------------------------------------------------
// CDNA5 (gfx1250) flash-attention cross-attention gate.
// Matmuls: v_wmma_f32_16x16x32_f16 (wave32, f32 accum).
// V tiles staged into LDS via async global->LDS copies (double buffered,
// XOR-swizzled to kill bank conflicts), shared by the 4 waves of the block.
// ---------------------------------------------------------------------------

typedef __attribute__((ext_vector_type(16))) _Float16 v16h;
typedef __attribute__((ext_vector_type(8)))  float    v8f;
typedef __attribute__((ext_vector_type(4)))  int      i32x4;
typedef __attribute__((ext_vector_type(4)))  float    f32x4;

union HFrag {                 // one 16-bit WMMA operand (8 VGPRs / lane)
  v16h    h;
  unsigned w[8];
  i32x4   q[2];
};

__device__ __forceinline__ v8f v8f_zero() {
  v8f z = {0.f,0.f,0.f,0.f,0.f,0.f,0.f,0.f};
  return z;
}

__device__ __forceinline__ unsigned pk2(float a, float b) {
  union { _Float16 h[2]; unsigned u; } r;
  r.h[0] = (_Float16)a; r.h[1] = (_Float16)b;
  return r.u;
}

__device__ __forceinline__ v8f wmma16(const v16h& a, const v16h& b, v8f c) {
  return __builtin_amdgcn_wmma_f32_16x16x32_f16(false, a, false, b, (short)0, c,
                                                false, false);
}

// ---- gfx1250 async global->LDS copy (guarded; sync fallback) --------------
// Builtin signature (from hipcc diagnostic): param0 = v4i in global AS,
// param1 = v4i in LDS AS, then imm offset, imm cpol.
typedef __attribute__((address_space(1))) i32x4 gi128;
typedef __attribute__((address_space(3))) i32x4 li128;

#if __has_builtin(__builtin_amdgcn_global_load_async_to_lds_b128)
#define HAVE_ASYNC_LDS 1
__device__ __forceinline__ void cp16_async(const _Float16* g, _Float16* l) {
  __builtin_amdgcn_global_load_async_to_lds_b128((gi128*)g, (li128*)l, 0, 0);
}
#else
__device__ __forceinline__ void cp16_async(const _Float16* g, _Float16* l) {
  *(i32x4*)l = *(const i32x4*)g;            // sync fallback: load + ds_store
}
#endif

#if __has_builtin(__builtin_amdgcn_s_wait_asynccnt)
#define ASYNC_WAIT(n) __builtin_amdgcn_s_wait_asynccnt(n)
#elif defined(HAVE_ASYNC_LDS)
#define ASYNC_WAIT(n) asm volatile("s_wait_asynccnt %0" ::"i"(n))
#else
#define ASYNC_WAIT(n)
#endif

// ---------------------------------------------------------------------------
// Projection: out[o,n] = sum_c W[o,c] * X[b,c,n] + bias[o], per batch b.
// PIXEL_MAJOR: out[b,n,o] f16 (q,k);  else out[b,o,n] f16 (v).
// ---------------------------------------------------------------------------
template<int COUT, bool PIXEL_MAJOR>
__global__ void __launch_bounds__(128)
proj_kernel(const float* __restrict__ X, const float* __restrict__ W,
            const float* __restrict__ bias, _Float16* __restrict__ out,
            int Bn, int Cin, int N)
{
  constexpr int OT = COUT / 16;
  const int tid  = threadIdx.x;
  const int lane = tid & 31;
  const int wave = tid >> 5;
  const int col  = lane & 15;
  const int h    = lane >> 4;
  const int wid  = blockIdx.x * 4 + wave;
  const int ntiles = N >> 4;
  const int b   = wid / ntiles;
  const int nt  = wid % ntiles;
  if (b >= Bn) return;
  const int n = nt * 16 + col;

  v8f acc[OT];
#pragma unroll
  for (int i = 0; i < OT; ++i) acc[i] = v8f_zero();

  for (int c0 = 0; c0 < Cin; c0 += 32) {
    HFrag bf;   // B frag (32x16): lanes 0-15 K=c0+0..15, lanes 16-31 K=c0+16..31
    const float* xp = X + ((size_t)b * Cin + (size_t)(c0 + 16 * h)) * N + n;
#pragma unroll
    for (int v = 0; v < 8; ++v)
      bf.w[v] = pk2(xp[(size_t)(2 * v) * N], xp[(size_t)(2 * v + 1) * N]);

#pragma unroll
    for (int ot = 0; ot < OT; ++ot) {
      const float* wp = W + (size_t)(ot * 16 + col) * Cin + c0 + 16 * h;
      f32x4 a0 = *(const f32x4*)(wp);
      f32x4 a1 = *(const f32x4*)(wp + 4);
      f32x4 a2 = *(const f32x4*)(wp + 8);
      f32x4 a3 = *(const f32x4*)(wp + 12);
      HFrag af;
      af.w[0] = pk2(a0.x, a0.y); af.w[1] = pk2(a0.z, a0.w);
      af.w[2] = pk2(a1.x, a1.y); af.w[3] = pk2(a1.z, a1.w);
      af.w[4] = pk2(a2.x, a2.y); af.w[5] = pk2(a2.z, a2.w);
      af.w[6] = pk2(a3.x, a3.y); af.w[7] = pk2(a3.z, a3.w);
      acc[ot] = wmma16(af.h, bf.h, acc[ot]);
    }
  }

#pragma unroll
  for (int ot = 0; ot < OT; ++ot) {
    const float* bp = bias + ot * 16 + 8 * h;
#pragma unroll
    for (int r = 0; r < 8; ++r) acc[ot][r] += bp[r];
    if (PIXEL_MAJOR) {
      _Float16* op = out + ((size_t)b * N + n) * COUT + ot * 16 + 8 * h;
      i32x4 st;
      st.x = pk2(acc[ot][0], acc[ot][1]);
      st.y = pk2(acc[ot][2], acc[ot][3]);
      st.z = pk2(acc[ot][4], acc[ot][5]);
      st.w = pk2(acc[ot][6], acc[ot][7]);
      *(i32x4*)op = st;
    } else {
#pragma unroll
      for (int r = 0; r < 8; ++r) {
        int o = ot * 16 + 8 * h + r;
        out[((size_t)b * COUT + o) * N + n] = (_Float16)acc[ot][r];
      }
    }
  }
}

// ---------------------------------------------------------------------------
// Flash attention (transposed-S). Block = 4 waves, 64 query columns, all 256
// output channels; j streamed in 64s. V staged in LDS (double buffered).
// LDS V layout: row c (0..255) of 8 16B-chunks, chunk index ^= (c&7).
// ---------------------------------------------------------------------------
__global__ void __launch_bounds__(128)
attn_kernel(const _Float16* __restrict__ qw, const _Float16* __restrict__ kw,
            const _Float16* __restrict__ vw, const float* __restrict__ skip,
            const float* __restrict__ gamma, float* __restrict__ out,
            int Bn, int N)
{
  constexpr int CS = 256, D = 64, CT = CS / 16;
  __shared__ _Float16 vbuf[2][CS * 64];      // 2 x 32KB

  const int tid  = threadIdx.x;
  const int lane = tid & 31;
  const int wave = tid >> 5;
  const int col  = lane & 15;
  const int h    = lane >> 4;
  const int itiles = N >> 6;
  const int b  = blockIdx.x / itiles;
  const int it = blockIdx.x % itiles;
  if (b >= Bn) return;
  const int i = it * 64 + wave * 16 + col;

  const _Float16* kbase = kw + (size_t)b * N * D;
  const _Float16* vbase = vw + (size_t)b * CS * N;

  // q as resident B-operand (32x16, K=d)
  HFrag qb0, qb1;
  {
    const _Float16* qp = qw + ((size_t)b * N + i) * D + h * 16;
    qb0.q[0] = *(const i32x4*)(qp);
    qb0.q[1] = *(const i32x4*)(qp + 8);
    qb1.q[0] = *(const i32x4*)(qp + 32);
    qb1.q[1] = *(const i32x4*)(qp + 40);
  }

  v8f O[CT];
#pragma unroll
  for (int c = 0; c < CT; ++c) O[c] = v8f_zero();
  float m = -3.0e38f, lsum = 0.f;

  // stage V tile for j0 into vbuf[buf]: 2048 16B chunks, 512 per wave
  auto stage_v = [&](int buf, int j0) {
#pragma unroll
    for (int itr = 0; itr < 16; ++itr) {
      int t    = (wave * 16 + itr) * 32 + lane;
      int row  = t >> 3;
      int part = t & 7;
      const _Float16* g = vbase + (size_t)row * N + j0 + part * 8;
      _Float16* l = &vbuf[buf][row * 64 + ((part ^ (row & 7)) * 8)];
      cp16_async(g, l);
    }
  };

  int cur = 0;
  stage_v(cur, 0);

  for (int j0 = 0; j0 < N; j0 += 64) {
    const bool more = (j0 + 64) < N;
    if (more) stage_v(cur ^ 1, j0 + 64);
    if (more) { ASYNC_WAIT(16); } else { ASYNC_WAIT(0); }
    __syncthreads();                         // vbuf[cur] ready for all waves

    // ---- S' = K * q (K from global, paired prefetch) ----
    v8f S[4];
#pragma unroll
    for (int jp = 0; jp < 2; ++jp) {
      HFrag ka[2][2];
#pragma unroll
      for (int u = 0; u < 2; ++u) {
        const int js = 2 * jp + u;
        const _Float16* kp = kbase + (size_t)(j0 + js * 16 + col) * D + h * 16;
        ka[u][0].q[0] = *(const i32x4*)(kp);
        ka[u][0].q[1] = *(const i32x4*)(kp + 8);
        ka[u][1].q[0] = *(const i32x4*)(kp + 32);
        ka[u][1].q[1] = *(const i32x4*)(kp + 40);
      }
#pragma unroll
      for (int u = 0; u < 2; ++u) {
        v8f s = wmma16(ka[u][0].h, qb0.h, v8f_zero());
        s = wmma16(ka[u][1].h, qb1.h, s);
        S[2 * jp + u] = s;
      }
    }

    // ---- online softmax over j ----
    float mloc = -3.0e38f;
#pragma unroll
    for (int js = 0; js < 4; ++js)
#pragma unroll
      for (int r = 0; r < 8; ++r) mloc = fmaxf(mloc, S[js][r]);
    mloc = fmaxf(mloc, __shfl_xor(mloc, 16, 32));
    float mnew  = fmaxf(m, mloc);
    float alpha = __expf(m - mnew);
    float rs = 0.f;
#pragma unroll
    for (int js = 0; js < 4; ++js)
#pragma unroll
      for (int r = 0; r < 8; ++r) {
        float p = __expf(S[js][r] - mnew);
        S[js][r] = p;
        rs += p;
      }
    rs += __shfl_xor(rs, 16, 32);
    lsum = lsum * alpha + rs;
    m = mnew;
#pragma unroll
    for (int c = 0; c < CT; ++c) O[c] *= alpha;

    // ---- O += V * P'  (V from LDS, P' via shfl_xor(16) repack) ----
#pragma unroll
    for (int jt = 0; jt < 2; ++jt) {
      HFrag pb;
#pragma unroll
      for (int v = 0; v < 4; ++v) {
        unsigned t1 = pk2(S[2 * jt][2 * v],     S[2 * jt][2 * v + 1]);
        unsigned t2 = pk2(S[2 * jt + 1][2 * v], S[2 * jt + 1][2 * v + 1]);
        unsigned s1 = (unsigned)__shfl_xor((int)t1, 16, 32);
        unsigned s2 = (unsigned)__shfl_xor((int)t2, 16, 32);
        pb.w[v]     = (h == 0) ? t1 : s2;
        pb.w[v + 4] = (h == 0) ? s1 : t2;
      }
      const int p0 = (4 * jt + 2 * h);       // 16B chunk indices in LDS row
      const int p1 = (4 * jt + 2 * h + 1);
#pragma unroll
      for (int ct = 0; ct < CT; ++ct) {
        const int row = ct * 16 + col;
        const int key = col & 7;             // (row & 7) == (col & 7)
        HFrag va;
        va.q[0] = *(const i32x4*)&vbuf[cur][row * 64 + ((p0 ^ key) * 8)];
        va.q[1] = *(const i32x4*)&vbuf[cur][row * 64 + ((p1 ^ key) * 8)];
        O[ct] = wmma16(va.h, pb.h, O[ct]);
      }
    }

    __syncthreads();                         // done reading vbuf[cur]
    cur ^= 1;
  }

  // ---- epilogue: out = gamma * O / l + skip ----
  const float scale = gamma[0] / lsum;
#pragma unroll
  for (int ct = 0; ct < CT; ++ct) {
#pragma unroll
    for (int r = 0; r < 8; ++r) {
      int c = ct * 16 + 8 * h + r;
      size_t idx = ((size_t)b * CS + c) * N + i;
      out[idx] = O[ct][r] * scale + skip[idx];
    }
  }
}

// ---------------------------------------------------------------------------
extern "C" void kernel_launch(void* const* d_in, const int* in_sizes, int n_in,
                              void* d_out, int out_size, void* d_ws, size_t ws_size,
                              hipStream_t stream) {
  const float* gate  = (const float*)d_in[0];
  const float* skip  = (const float*)d_in[1];
  const float* Wq    = (const float*)d_in[2];
  const float* bq    = (const float*)d_in[3];
  const float* Wk    = (const float*)d_in[4];
  const float* bk    = (const float*)d_in[5];
  const float* Wv    = (const float*)d_in[6];
  const float* bv    = (const float*)d_in[7];
  const float* gamma = (const float*)d_in[8];
  float* out = (float*)d_out;

  const int CG = 512, CS = 256, INTER = 64;
  const int B = in_sizes[0] / (CG * 64 * 64);
  const int N = in_sizes[1] / (B * CS);

  char* ws = (char*)d_ws;
  const size_t qk_bytes = (size_t)B * N * INTER * sizeof(_Float16);
  _Float16* q_ws = (_Float16*)(ws);
  _Float16* k_ws = (_Float16*)(ws + qk_bytes);
  _Float16* v_ws = (_Float16*)(ws + 2 * qk_bytes);

  {
    const int waves = B * (N / 16);
    dim3 grid(waves / 4), block(128);
    proj_kernel<64, true ><<<grid, block, 0, stream>>>(gate, Wq, bq, q_ws, B, CG, N);
    proj_kernel<64, true ><<<grid, block, 0, stream>>>(skip, Wk, bk, k_ws, B, CS, N);
    proj_kernel<256, false><<<grid, block, 0, stream>>>(skip, Wv, bv, v_ws, B, CS, N);
  }
  {
    dim3 grid(B * (N / 64)), block(128);
    attn_kernel<<<grid, block, 0, stream>>>(q_ws, k_ws, v_ws, skip, gamma, out, B, N);
  }
}
